// Multihead_Attention_Main_57801669869660
// MI455X (gfx1250) — matrix operand order repaired and checked by reading, not compiled, not run on hardware
//
#include <hip/hip_runtime.h>
#include <hip/hip_bf16.h>
#include <stdint.h>

// Multi-head causal attention, B=4 S=2048 D=1024 H=16 HD=64, bf16 WMMA pipeline.
#define BB 4
#define SS 2048
#define DD 1024
#define HH 16
#define HDIM 64

typedef __bf16 bf16_t;
typedef __attribute__((ext_vector_type(16))) __bf16 v16bf;
typedef __attribute__((ext_vector_type(8)))  __bf16 v8bf;
typedef __attribute__((ext_vector_type(4)))  __bf16 v4bf;
typedef __attribute__((ext_vector_type(8)))  float  v8f;
typedef __attribute__((ext_vector_type(4)))  float  v4f;

__device__ __forceinline__ bf16_t f2bf(float f) {
  union { float f; uint32_t u; } c; c.f = f;
  uint32_t u = c.u + 0x7FFFu + ((c.u >> 16) & 1u);   // round-to-nearest-even
  uint16_t h = (uint16_t)(u >> 16);
  bf16_t r; __builtin_memcpy(&r, &h, 2); return r;
}

// Build a 16-element bf16 fragment from two contiguous 16-byte chunks.
__device__ __forceinline__ v16bf frag16(const bf16_t* p0, const bf16_t* p1) {
  v8bf a = *(const v8bf*)p0;
  v8bf b = *(const v8bf*)p1;
  v16bf r;
#pragma unroll
  for (int i = 0; i < 8; ++i) { r[i] = a[i]; r[i + 8] = b[i]; }
  return r;
}

// 4-wide f32 -> bf16 convert (b128 read, b64 write).
__global__ void cvt_f32_bf16(const float* __restrict__ in, bf16_t* __restrict__ out, int n4) {
  int i = blockIdx.x * blockDim.x + threadIdx.x;
  if (i < n4) {
    v4f v = *(const v4f*)(in + (size_t)i * 4);
    v4bf o;
#pragma unroll
    for (int j = 0; j < 4; ++j) o[j] = f2bf(v[j]);
    *(v4bf*)(out + (size_t)i * 4) = o;
  }
}

// C[M,N] = A[M,K] @ W[K,N]; A,W bf16; output bf16 (Cb) or f32+bias (Cf).
// Block: 256 threads (8 waves). Block tile 128x128, K-chunks of 32, LDS ping-pong.
// Wave grid 4x2: wave computes rows [wm*32, +32), cols [wn*64, +64) of the tile.
__global__ __launch_bounds__(256) void gemm_bf16_wmma(
    const bf16_t* __restrict__ A, const bf16_t* __restrict__ W,
    bf16_t* __restrict__ Cb, float* __restrict__ Cf, const float* __restrict__ bias,
    int M, int N, int K)
{
  __shared__ bf16_t As[2][128][40];    // row stride 80B  (16B-aligned)
  __shared__ bf16_t Bs[2][32][136];    // row stride 272B (16B-aligned)

  const int t = threadIdx.x;
  const int wave = t >> 5, lane = t & 31;
  const int hi = lane >> 4, lo = lane & 15;
  const int wm = wave >> 1, wn = wave & 1;
  const int bm = blockIdx.y, bn = blockIdx.x;
  const int kb = hi * 8;

  v8f acc[2][4] = {{{}, {}, {}, {}}, {{}, {}, {}, {}}};

  const int arow = t >> 1, aseg = (t & 1) << 4;   // A tile: 128x32, 16 elems/thread
  const int brow = t >> 3, bseg = (t & 7) << 4;   // W tile: 32x128, 16 elems/thread
  const int nK = K >> 5;

  // stage chunk 0 into buffer 0
  {
    const bf16_t* g = &A[(size_t)(bm * 128 + arow) * K + aseg];
    *(v8bf*)&As[0][arow][aseg]     = *(const v8bf*)g;
    *(v8bf*)&As[0][arow][aseg + 8] = *(const v8bf*)(g + 8);
    const bf16_t* w = &W[(size_t)brow * N + bn * 128 + bseg];
    *(v8bf*)&Bs[0][brow][bseg]     = *(const v8bf*)w;
    *(v8bf*)&Bs[0][brow][bseg + 8] = *(const v8bf*)(w + 8);
  }

  for (int kt = 0; kt < nK; ++kt) {
    const int cur = kt & 1;
    __syncthreads();                       // buffer `cur` ready; buffer `cur^1` free
    if (kt + 1 < nK) {                     // stage next chunk into the other buffer
      const bf16_t* g = &A[(size_t)(bm * 128 + arow) * K + (kt + 1) * 32 + aseg];
      *(v8bf*)&As[cur ^ 1][arow][aseg]     = *(const v8bf*)g;
      *(v8bf*)&As[cur ^ 1][arow][aseg + 8] = *(const v8bf*)(g + 8);
      const bf16_t* w = &W[(size_t)((kt + 1) * 32 + brow) * N + bn * 128 + bseg];
      *(v8bf*)&Bs[cur ^ 1][brow][bseg]     = *(const v8bf*)w;
      *(v8bf*)&Bs[cur ^ 1][brow][bseg + 8] = *(const v8bf*)(w + 8);
    }

    v16bf aF[2];
#pragma unroll
    for (int mt = 0; mt < 2; ++mt)
      aF[mt] = frag16(&As[cur][wm * 32 + mt * 16 + lo][kb],
                      &As[cur][wm * 32 + mt * 16 + lo][kb + 16]);
#pragma unroll
    for (int nt = 0; nt < 4; ++nt) {
      v16bf b;
#pragma unroll
      for (int s = 0; s < 16; ++s)
        b[s] = Bs[cur][s < 8 ? kb + s : 16 + kb + (s - 8)][wn * 64 + nt * 16 + lo];
#pragma unroll
      for (int mt = 0; mt < 2; ++mt)
        acc[mt][nt] = __builtin_amdgcn_wmma_f32_16x16x32_bf16(
            false, aF[mt], false, b, (short)0, acc[mt][nt], false, false);
    }
  }

#pragma unroll
  for (int mt = 0; mt < 2; ++mt)
#pragma unroll
    for (int nt = 0; nt < 4; ++nt)
#pragma unroll
      for (int r = 0; r < 8; ++r) {
        int row = bm * 128 + wm * 32 + mt * 16 + r + hi * 8;
        int col = bn * 128 + wn * 64 + nt * 16 + lo;
        if (Cf) Cf[(size_t)row * N + col] = acc[mt][nt][r] + (bias ? bias[col] : 0.0f);
        else    Cb[(size_t)row * N + col] = f2bf(acc[mt][nt][r]);
      }
}

// Flash-style causal attention. Grid (S/64, H, B), block 128 (4 waves).
// Q,K,V,ctx laid out [B, S, H*64] bf16 (head h = columns h*64..h*64+63).
__global__ __launch_bounds__(128) void attn_wmma(
    const bf16_t* __restrict__ Q, const bf16_t* __restrict__ Kg,
    const bf16_t* __restrict__ V, bf16_t* __restrict__ ctx)
{
  __shared__ bf16_t Ks[64][72];        // transposed: [d][key]
  __shared__ bf16_t Vs[64][72];        // [key][d]
  __shared__ bf16_t Ps[4][16][72];     // per-wave P staging (C-layout -> A-layout)

  const int t = threadIdx.x;
  const int wave = t >> 5, lane = t & 31;
  const int hi = lane >> 4, lo = lane & 15;
  const int qb = blockIdx.x, h = blockIdx.y, b = blockIdx.z;
  const size_t hbase = (size_t)h * HDIM;

  const bf16_t* Qp = Q  + (size_t)b * SS * DD;
  const bf16_t* Kp = Kg + (size_t)b * SS * DD;
  const bf16_t* Vp = V  + (size_t)b * SS * DD;

  // Per-wave Q fragments (rows qb*64 + wave*16 + lo), d-chunks of 32.
  const int kb = hi * 8;
  const bf16_t* qrow = Qp + (size_t)(qb * 64 + wave * 16 + lo) * DD + hbase;
  v16bf aQ0 = frag16(qrow + kb,      qrow + kb + 16);
  v16bf aQ1 = frag16(qrow + 32 + kb, qrow + 32 + kb + 16);

  v8f acc[4] = {{}, {}, {}, {}};
  float mrow[8], lrow[8];
#pragma unroll
  for (int r = 0; r < 8; ++r) { mrow[r] = -__builtin_inff(); lrow[r] = 0.0f; }
  const float LOG2E = 1.44269504088896f;

  for (int jb = 0; jb <= qb; ++jb) {
    {   // cooperative loads: K transposed, V natural; 32 elems/thread each
      const int key = t >> 1, ds = (t & 1) << 5;
      const bf16_t* gk = Kp + (size_t)(jb * 64 + key) * DD + hbase + ds;
      v8bf k0 = *(const v8bf*)(gk);
      v8bf k1 = *(const v8bf*)(gk + 8);
      v8bf k2 = *(const v8bf*)(gk + 16);
      v8bf k3 = *(const v8bf*)(gk + 24);
#pragma unroll
      for (int i = 0; i < 8; ++i) {
        Ks[ds + i     ][key] = k0[i];
        Ks[ds + i + 8 ][key] = k1[i];
        Ks[ds + i + 16][key] = k2[i];
        Ks[ds + i + 24][key] = k3[i];
      }
      const bf16_t* gv = Vp + (size_t)(jb * 64 + key) * DD + hbase + ds;
      *(v8bf*)&Vs[key][ds]      = *(const v8bf*)(gv);
      *(v8bf*)&Vs[key][ds + 8]  = *(const v8bf*)(gv + 8);
      *(v8bf*)&Vs[key][ds + 16] = *(const v8bf*)(gv + 16);
      *(v8bf*)&Vs[key][ds + 24] = *(const v8bf*)(gv + 24);
    }
    __syncthreads();

    // S = Q @ K^T  (16 queries x 64 keys per wave)
    v8f sc[4] = {{}, {}, {}, {}};
#pragma unroll
    for (int nt = 0; nt < 4; ++nt) {
      v16bf bk0, bk1;
#pragma unroll
      for (int s = 0; s < 16; ++s) {
        int kk = s < 8 ? kb + s : 16 + kb + (s - 8);
        bk0[s] = Ks[kk][nt * 16 + lo];
        bk1[s] = Ks[32 + kk][nt * 16 + lo];
      }
      sc[nt] = __builtin_amdgcn_wmma_f32_16x16x32_bf16(
          false, aQ0, false, bk0, (short)0, sc[nt], false, false);

      sc[nt] = __builtin_amdgcn_wmma_f32_16x16x32_bf16(
          false, aQ1, false, bk1, (short)0, sc[nt], false, false);
    }

    const bool diag = (jb == qb);
#pragma unroll
    for (int nt = 0; nt < 4; ++nt)
#pragma unroll
      for (int r = 0; r < 8; ++r) {
        float v = sc[nt][r] * 0.125f;               // 1/sqrt(64)
        if (diag) {
          int qloc = wave * 16 + r + hi * 8;
          int kloc = nt * 16 + lo;
          if (kloc > qloc) v = -__builtin_inff();   // causal mask
        }
        sc[nt][r] = v;
      }

    // online softmax: row lives across a 16-lane half-group
    float mnew[8];
#pragma unroll
    for (int r = 0; r < 8; ++r) {
      float v = fmaxf(fmaxf(sc[0][r], sc[1][r]), fmaxf(sc[2][r], sc[3][r]));
#pragma unroll
      for (int off = 1; off < 16; off <<= 1) v = fmaxf(v, __shfl_xor(v, off, 32));
      mnew[r] = fmaxf(mrow[r], v);
    }
#pragma unroll
    for (int nt = 0; nt < 4; ++nt)
#pragma unroll
      for (int r = 0; r < 8; ++r)
        sc[nt][r] = exp2f((sc[nt][r] - mnew[r]) * LOG2E);
#pragma unroll
    for (int r = 0; r < 8; ++r) {
      float s = sc[0][r] + sc[1][r] + sc[2][r] + sc[3][r];
#pragma unroll
      for (int off = 1; off < 16; off <<= 1) s += __shfl_xor(s, off, 32);
      float esc = exp2f((mrow[r] - mnew[r]) * LOG2E);
      lrow[r] = lrow[r] * esc + s;
      mrow[r] = mnew[r];
#pragma unroll
      for (int dt = 0; dt < 4; ++dt) acc[dt][r] *= esc;
    }

    // stage P (C-layout) into wave-private LDS, re-read in A-layout
#pragma unroll
    for (int nt = 0; nt < 4; ++nt)
#pragma unroll
      for (int r = 0; r < 8; ++r)
        Ps[wave][r + hi * 8][nt * 16 + lo] = f2bf(sc[nt][r]);

    // O += P @ V
#pragma unroll
    for (int kc = 0; kc < 2; ++kc) {
      v16bf aP = frag16(&Ps[wave][lo][kc * 32 + kb], &Ps[wave][lo][kc * 32 + kb + 16]);
#pragma unroll
      for (int dt = 0; dt < 4; ++dt) {
        v16bf bv;
#pragma unroll
        for (int s = 0; s < 16; ++s)
          bv[s] = Vs[kc * 32 + (s < 8 ? kb + s : 16 + kb + (s - 8))][dt * 16 + lo];
        acc[dt] = __builtin_amdgcn_wmma_f32_16x16x32_bf16(
            false, aP, false, bv, (short)0, acc[dt], false, false);
      }
    }
    __syncthreads();
  }

  // O /= l, write context (bf16, head-contiguous layout)
  bf16_t* crow = ctx + (size_t)b * SS * DD + hbase;
#pragma unroll
  for (int r = 0; r < 8; ++r) {
    float inv = 1.0f / lrow[r];
    int q = qb * 64 + wave * 16 + r + hi * 8;
#pragma unroll
    for (int dt = 0; dt < 4; ++dt)
      crow[(size_t)q * DD + dt * 16 + lo] = f2bf(acc[dt][r] * inv);
  }
}

extern "C" void kernel_launch(void* const* d_in, const int* in_sizes, int n_in,
                              void* d_out, int out_size, void* d_ws, size_t ws_size,
                              hipStream_t stream) {
  (void)in_sizes; (void)n_in; (void)out_size; (void)ws_size;
  const float* x  = (const float*)d_in[0];
  const float* Wq = (const float*)d_in[1];
  const float* Wk = (const float*)d_in[2];
  const float* Wv = (const float*)d_in[3];
  const float* Wo = (const float*)d_in[4];
  const float* bo = (const float*)d_in[5];
  float* out = (float*)d_out;

  const int M = BB * SS;                 // 8192
  const size_t nx = (size_t)M * DD;      // 8.39M elems
  const size_t nw = (size_t)DD * DD;     // 1.05M elems

  bf16_t* p = (bf16_t*)d_ws;
  bf16_t* xb  = p; p += nx;
  bf16_t* wqb = p; p += nw;
  bf16_t* wkb = p; p += nw;
  bf16_t* wvb = p; p += nw;
  bf16_t* wob = p; p += nw;
  bf16_t* Qb  = p; p += nx;
  bf16_t* Kb  = p; p += nx;
  bf16_t* Vb  = p; p += nx;
  bf16_t* Cb  = p; p += nx;

  const int CT = 256;
  const int nx4 = (int)(nx / 4), nw4 = (int)(nw / 4);
  cvt_f32_bf16<<<dim3((nx4 + CT - 1) / CT), CT, 0, stream>>>(x,  xb,  nx4);
  cvt_f32_bf16<<<dim3((nw4 + CT - 1) / CT), CT, 0, stream>>>(Wq, wqb, nw4);
  cvt_f32_bf16<<<dim3((nw4 + CT - 1) / CT), CT, 0, stream>>>(Wk, wkb, nw4);
  cvt_f32_bf16<<<dim3((nw4 + CT - 1) / CT), CT, 0, stream>>>(Wv, wvb, nw4);
  cvt_f32_bf16<<<dim3((nw4 + CT - 1) / CT), CT, 0, stream>>>(Wo, wob, nw4);

  dim3 gg(DD / 128, M / 128);            // 8 x 64
  gemm_bf16_wmma<<<gg, 256, 0, stream>>>(xb, wqb, Qb, nullptr, nullptr, M, DD, DD);
  gemm_bf16_wmma<<<gg, 256, 0, stream>>>(xb, wkb, Kb, nullptr, nullptr, M, DD, DD);
  gemm_bf16_wmma<<<gg, 256, 0, stream>>>(xb, wvb, Vb, nullptr, nullptr, M, DD, DD);

  dim3 ga(SS / 64, HH, BB);              // 32 x 16 x 4
  attn_wmma<<<ga, 128, 0, stream>>>(Qb, Kb, Vb, Cb);

  gemm_bf16_wmma<<<gg, 256, 0, stream>>>(Cb, wob, nullptr, out, bo, M, DD, DD);
}
